// GAttention_12335146074403
// MI455X (gfx1250) — matrix operand order repaired and checked by reading, not compiled, not run on hardware
//
#include <hip/hip_runtime.h>

typedef __bf16 bf16;
typedef bf16  v16bf __attribute__((ext_vector_type(16)));
typedef bf16  v8bf  __attribute__((ext_vector_type(8)));
typedef float v8f   __attribute__((ext_vector_type(8)));

#define NB    2
#define NH    12
#define SEQ   2048
#define HD    64
#define DIMC  768
#define QK_SCALE 0.125f   // 64^-0.5

static __device__ __forceinline__ v8f vzero8() {
    v8f z;
#pragma unroll
    for (int i = 0; i < 8; ++i) z[i] = 0.0f;
    return z;
}

static __device__ __forceinline__ v8f wmma_bf16(v16bf a, v16bf b, v8f c) {
    return __builtin_amdgcn_wmma_f32_16x16x32_bf16(false, a, false, b, (short)0, c, false, false);
}

// A fragment from bf16 row-major. p = row base + kb. Lane holds row (lane&15).
// halves j : K = kb + 16*hi + j
static __device__ __forceinline__ v16bf frag_a_bf16(const bf16* p, int hi) {
    v8bf lo = *(const v8bf*)(p + 16 * hi);
    v8bf hh = *(const v8bf*)(p + 16 * hi + 8);
    v16bf a;
#pragma unroll
    for (int j = 0; j < 8; ++j) { a[j] = lo[j]; a[j + 8] = hh[j]; }
    return a;
}

// B fragment from bf16 row-major W[N][K] used as W^T. p = W-row base + kb.
// Lane holds column (lane&15) = W row; halves j : K = kb + 16*hi + j (one 32B load)
static __device__ __forceinline__ v16bf frag_b_bf16(const bf16* p, int hi) {
    return *(const v16bf*)(p + 16 * hi);
}

// Per-lane async copy of 16 bytes global -> LDS (ASYNCcnt-tracked).
// LDS operand = low 32 bits of the generic pointer (ISA 10.2: LDS aperture uses addr[31:0]).
static __device__ __forceinline__ void async_b128(const void* g, void* l) {
    unsigned loff = (unsigned)(size_t)l;
    unsigned long long ga = (unsigned long long)g;
    asm volatile("global_load_async_to_lds_b128 %0, %1, off"
                 :: "v"(loff), "v"(ga) : "memory");
}

// ---------------------------------------------------------------------------
// Prep kernels: fp32 -> bf16 conversion (4 elems/thread) and V transpose.
// ---------------------------------------------------------------------------
__global__ void cvt_kernel(const float* __restrict__ in, bf16* __restrict__ out) {
    size_t i = ((size_t)blockIdx.x * blockDim.x + threadIdx.x) * 4;
    float4 f = *(const float4*)(in + i);
    union { bf16 h[4]; uint2 u; } t;
    t.h[0] = (bf16)f.x; t.h[1] = (bf16)f.y; t.h[2] = (bf16)f.z; t.h[3] = (bf16)f.w;
    *(uint2*)(out + i) = t.u;
}

__global__ void vt_kernel(const float* __restrict__ v, bf16* __restrict__ vt) {
    size_t idx = (size_t)blockIdx.x * blockDim.x + threadIdx.x;
    size_t bh  = idx / ((size_t)SEQ * HD);
    size_t rem = idx % ((size_t)SEQ * HD);
    size_t mm  = rem / HD;
    size_t dd  = rem % HD;
    vt[(bh * HD + dd) * SEQ + mm] = (bf16)v[idx];
}

// ---------------------------------------------------------------------------
// Kernel: Q = (X @ Wqkv^T) * scale -> bf16 [B,H,N,d]
// Wave computes a 16x32 output block (2 WMMA accumulators sharing the A frag).
// ---------------------------------------------------------------------------
__global__ void qproj_kernel(const bf16* __restrict__ xb, const bf16* __restrict__ wb,
                             bf16* __restrict__ q_ws) {
    int wave = threadIdx.x >> 5, lane = threadIdx.x & 31;
    int hi = lane >> 4, lr = lane & 15;
    int rowbase = (blockIdx.y * 4 + wave) * 16;
    int colbase = blockIdx.x * 32;
    const bf16* arow  = xb + (size_t)(rowbase + lr) * DIMC;
    const bf16* brow0 = wb + (size_t)(colbase + lr) * DIMC;
    const bf16* brow1 = wb + (size_t)(colbase + 16 + lr) * DIMC;
    v8f acc0 = vzero8(), acc1 = vzero8();
    for (int kb = 0; kb < DIMC; kb += 32) {
        v16bf a  = frag_a_bf16(arow + kb, hi);
        v16bf b0 = frag_b_bf16(brow0 + kb, hi);
        v16bf b1 = frag_b_bf16(brow1 + kb, hi);
        acc0 = wmma_bf16(a, b0, acc0);
        acc1 = wmma_bf16(a, b1, acc1);
    }
    int col0 = colbase + lr, col1 = colbase + 16 + lr;
    int h0 = col0 >> 6, d0 = col0 & 63;
    int h1 = col1 >> 6, d1 = col1 & 63;
#pragma unroll
    for (int r = 0; r < 8; ++r) {
        int row = rowbase + r + 8 * hi;        // row in [0, B*N)
        int bb = row >> 11, n = row & (SEQ - 1);
        q_ws[(((size_t)bb * NH + h0) * SEQ + n) * HD + d0] = (bf16)(acc0[r] * QK_SCALE);
        q_ws[(((size_t)bb * NH + h1) * SEQ + n) * HD + d1] = (bf16)(acc1[r] * QK_SCALE);
    }
}

// ---------------------------------------------------------------------------
// Flash attention. Block = 4 waves, one (b,h), 64 query rows.
// Double-buffered K/Vt tiles staged by GLOBAL_LOAD_ASYNC_TO_LDS_B128,
// prefetching step+1 while computing step (one barrier per step).
// ---------------------------------------------------------------------------
__global__ void attn_kernel(const bf16* __restrict__ q_ws, const bf16* __restrict__ kmat,
                            const bf16* __restrict__ vt_ws, bf16* __restrict__ attn_ws) {
    __shared__ __align__(32) bf16 Kt[2][32][80];     // [buf][m_local][d]
    __shared__ __align__(32) bf16 Vt[2][64][48];     // [buf][d][m_local]
    __shared__ __align__(32) bf16 Pt[4][16][48];     // per-wave P tile [row][col]

    int tid = threadIdx.x;
    int wave = tid >> 5, lane = tid & 31, hi = lane >> 4, lr = lane & 15;
    int bh = blockIdx.x;
    int qbase = (blockIdx.y * 4 + wave) * 16;

    const bf16* qrow = q_ws + ((size_t)bh * SEQ + qbase + lr) * HD;
    v16bf aq0 = frag_a_bf16(qrow, hi);        // d = 0..31  (pre-scaled)
    v16bf aq1 = frag_a_bf16(qrow + 32, hi);   // d = 32..63

    float m_i[8], l_i[8];
#pragma unroll
    for (int r = 0; r < 8; ++r) { m_i[r] = -1e30f; l_i[r] = 0.0f; }
    v8f o0 = vzero8(), o1 = vzero8(), o2 = vzero8(), o3 = vzero8();

    const bf16* kbp = kmat + (size_t)bh * SEQ * HD;
    const bf16* vtb = vt_ws + (size_t)bh * HD * SEQ;

    int km = tid >> 2, kd = (tid & 3) * 16;    // K tile: 32 rows x 64 d, 32B/thread
    int vd = tid >> 1, vm = (tid & 1) * 16;    // Vt tile: 64 rows x 32 m, 32B/thread

#define STAGE(buf, mt) do {                                                  \
        const bf16* kp_ = kbp + (size_t)((mt) + km) * HD + kd;               \
        async_b128(kp_,     &Kt[buf][km][kd]);                               \
        async_b128(kp_ + 8, &Kt[buf][km][kd + 8]);                           \
        const bf16* vp_ = vtb + (size_t)vd * SEQ + (mt) + vm;                \
        async_b128(vp_,     &Vt[buf][vd][vm]);                               \
        async_b128(vp_ + 8, &Vt[buf][vd][vm + 8]);                           \
    } while (0)

    STAGE(0, 0);
    const int NSTEP = SEQ / 32;
    for (int step = 0; step < NSTEP; ++step) {
        int buf = step & 1;
        asm volatile("s_wait_asynccnt 0" ::: "memory");
        __syncthreads();                        // tile[buf] ready; tile[buf^1] free
        if (step + 1 < NSTEP) STAGE(buf ^ 1, (step + 1) * 32);

        // --- S = Q K^T : two 16x16 tiles over 32 keys ---
        v8f s0 = vzero8(), s1 = vzero8();
        {
            v16bf b00 = *(const v16bf*)&Kt[buf][lr][hi * 16];
            v16bf b01 = *(const v16bf*)&Kt[buf][lr][32 + hi * 16];
            v16bf b10 = *(const v16bf*)&Kt[buf][16 + lr][hi * 16];
            v16bf b11 = *(const v16bf*)&Kt[buf][16 + lr][32 + hi * 16];
            s0 = wmma_bf16(aq0, b00, s0);
            s0 = wmma_bf16(aq1, b01, s0);
            s1 = wmma_bf16(aq0, b10, s1);
            s1 = wmma_bf16(aq1, b11, s1);
        }

        // --- online softmax: rows live per-lane (rows r+8*hi), reduce over 16 lanes ---
        float mnew[8];
#pragma unroll
        for (int r = 0; r < 8; ++r) mnew[r] = fmaxf(m_i[r], fmaxf(s0[r], s1[r]));
#pragma unroll
        for (int mask = 1; mask < 16; mask <<= 1) {
#pragma unroll
            for (int r = 0; r < 8; ++r) mnew[r] = fmaxf(mnew[r], __shfl_xor(mnew[r], mask, 32));
        }
        float p0v[8], p1v[8], rs[8], corr[8];
#pragma unroll
        for (int r = 0; r < 8; ++r) {
            p0v[r] = __expf(s0[r] - mnew[r]);
            p1v[r] = __expf(s1[r] - mnew[r]);
            rs[r] = p0v[r] + p1v[r];
            corr[r] = __expf(m_i[r] - mnew[r]);
        }
#pragma unroll
        for (int mask = 1; mask < 16; mask <<= 1) {
#pragma unroll
            for (int r = 0; r < 8; ++r) rs[r] += __shfl_xor(rs[r], mask, 32);
        }
#pragma unroll
        for (int r = 0; r < 8; ++r) {
            l_i[r] = l_i[r] * corr[r] + rs[r];
            m_i[r] = mnew[r];
            o0[r] *= corr[r]; o1[r] *= corr[r]; o2[r] *= corr[r]; o3[r] *= corr[r];
        }

        // --- P: C-layout -> A-layout via per-wave LDS tile ---
#pragma unroll
        for (int r = 0; r < 8; ++r) {
            Pt[wave][r + 8 * hi][lr]      = (bf16)p0v[r];
            Pt[wave][r + 8 * hi][16 + lr] = (bf16)p1v[r];
        }
        asm volatile("s_wait_dscnt 0" ::: "memory");
        v16bf ap;
        {
            v8bf lo = *(const v8bf*)&Pt[wave][lr][16 * hi];
            v8bf hh = *(const v8bf*)&Pt[wave][lr][16 * hi + 8];
#pragma unroll
            for (int j = 0; j < 8; ++j) { ap[j] = lo[j]; ap[j + 8] = hh[j]; }
        }

        // --- O += P V ---
        v16bf bv0 = *(const v16bf*)&Vt[buf][lr][hi * 16];
        v16bf bv1 = *(const v16bf*)&Vt[buf][16 + lr][hi * 16];
        v16bf bv2 = *(const v16bf*)&Vt[buf][32 + lr][hi * 16];
        v16bf bv3 = *(const v16bf*)&Vt[buf][48 + lr][hi * 16];
        o0 = wmma_bf16(ap, bv0, o0);
        o1 = wmma_bf16(ap, bv1, o1);
        o2 = wmma_bf16(ap, bv2, o2);
        o3 = wmma_bf16(ap, bv3, o3);
    }
#undef STAGE

    // --- normalize and store to attn_ws [B,N,768] bf16 ---
    int b = bh / NH, h = bh % NH;
#pragma unroll
    for (int r = 0; r < 8; ++r) {
        float inv = 1.0f / l_i[r];
        int q = qbase + r + 8 * hi;
        size_t base = ((size_t)b * SEQ + q) * DIMC + h * HD;
        attn_ws[base +      lr] = (bf16)(o0[r] * inv);
        attn_ws[base + 16 + lr] = (bf16)(o1[r] * inv);
        attn_ws[base + 32 + lr] = (bf16)(o2[r] * inv);
        attn_ws[base + 48 + lr] = (bf16)(o3[r] * inv);
    }
}

// ---------------------------------------------------------------------------
// Kernel: out = attn @ Wproj^T + bias (fp32 output), 16x32 block per wave.
// ---------------------------------------------------------------------------
__global__ void oproj_kernel(const bf16* __restrict__ attn, const bf16* __restrict__ wb,
                             const float* __restrict__ bias, float* __restrict__ out) {
    int wave = threadIdx.x >> 5, lane = threadIdx.x & 31;
    int hi = lane >> 4, lr = lane & 15;
    int rowbase = (blockIdx.y * 4 + wave) * 16;
    int colbase = blockIdx.x * 32;
    const bf16* arow  = attn + (size_t)(rowbase + lr) * DIMC;
    const bf16* brow0 = wb + (size_t)(colbase + lr) * DIMC;
    const bf16* brow1 = wb + (size_t)(colbase + 16 + lr) * DIMC;
    v8f acc0 = vzero8(), acc1 = vzero8();
    for (int kb = 0; kb < DIMC; kb += 32) {
        v16bf a  = frag_a_bf16(arow + kb, hi);
        v16bf b0 = frag_b_bf16(brow0 + kb, hi);
        v16bf b1 = frag_b_bf16(brow1 + kb, hi);
        acc0 = wmma_bf16(a, b0, acc0);
        acc1 = wmma_bf16(a, b1, acc1);
    }
    int col0 = colbase + lr, col1 = colbase + 16 + lr;
    float bv0 = bias[col0], bv1 = bias[col1];
#pragma unroll
    for (int r = 0; r < 8; ++r) {
        int row = rowbase + r + 8 * hi;
        out[(size_t)row * DIMC + col0] = acc0[r] + bv0;
        out[(size_t)row * DIMC + col1] = acc1[r] + bv1;
    }
}

extern "C" void kernel_launch(void* const* d_in, const int* in_sizes, int n_in,
                              void* d_out, int out_size, void* d_ws, size_t ws_size,
                              hipStream_t stream) {
    (void)in_sizes; (void)n_in; (void)out_size; (void)ws_size;
    const float* x      = (const float*)d_in[0];
    const float* k      = (const float*)d_in[1];
    const float* v      = (const float*)d_in[2];
    const float* w_qkv  = (const float*)d_in[3];
    const float* w_proj = (const float*)d_in[4];
    const float* b_proj = (const float*)d_in[5];
    float* out = (float*)d_out;

    char* ws = (char*)d_ws;
    const size_t BIG = (size_t)NB * NH * SEQ * HD * sizeof(bf16);   // 6,291,456 B
    const size_t WSEG = (size_t)DIMC * DIMC * sizeof(bf16);         // 1,179,648 B
    bf16* q_ws    = (bf16*)(ws);
    bf16* vt_ws   = (bf16*)(ws + BIG);
    bf16* attn_ws = (bf16*)(ws + 2 * BIG);
    bf16* xb      = (bf16*)(ws + 3 * BIG);
    bf16* kb      = (bf16*)(ws + 4 * BIG);
    bf16* wqkvb   = (bf16*)(ws + 5 * BIG);
    bf16* wprojb  = (bf16*)(ws + 5 * BIG + WSEG);

    const int NBIG = NB * NH * SEQ * HD;   // 3,145,728 (== NB*SEQ*DIMC)
    const int NW   = DIMC * DIMC;          // 589,824

    cvt_kernel<<<NBIG / 1024, 256, 0, stream>>>(x, xb);
    cvt_kernel<<<NBIG / 1024, 256, 0, stream>>>(k, kb);
    cvt_kernel<<<NW / 1024, 256, 0, stream>>>(w_qkv, wqkvb);
    cvt_kernel<<<NW / 1024, 256, 0, stream>>>(w_proj, wprojb);
    vt_kernel<<<NBIG / 256, 256, 0, stream>>>(v, vt_ws);

    qproj_kernel<<<dim3(DIMC / 32, (NB * SEQ) / 64), 128, 0, stream>>>(xb, wqkvb, q_ws);
    attn_kernel<<<dim3(NB * NH, SEQ / 64), 128, 0, stream>>>(q_ws, kb, vt_ws, attn_ws);
    oproj_kernel<<<dim3(DIMC / 32, (NB * SEQ) / 64), 128, 0, stream>>>(attn_ws, wprojb, b_proj, out);
}
